// TokenLabelOfflineEpisodic_22462678958124
// MI455X (gfx1250) — matrix operand order repaired and checked by reading, not compiled, not run on hardware
//
#include <hip/hip_runtime.h>
#include <math.h>

typedef _Float16 h16;
typedef __attribute__((ext_vector_type(16))) _Float16 v16h;
typedef __attribute__((ext_vector_type(8)))  float    v8f;
typedef __attribute__((ext_vector_type(4)))  int      v4i;

// pointee types in explicit address spaces for the async-copy builtin
typedef __attribute__((address_space(1))) v4i g_v4i;   // global
typedef __attribute__((address_space(3))) v4i l_v4i;   // LDS

#define B_   2
#define N_   5
#define K_   5
#define Q_   75
#define T_   196
#define C_   384

#define T_PAD     208          // 13 * 16
#define S_        (K_ * T_)    // 980
#define S_PAD     992          // 62 * 16
#define KB_       (C_ / 32)    // 12 k-steps of K=32
#define TT_TILES  13
#define ST_TILES  62
#define FRAG_HALFS 512         // 32 lanes * 16 f16 per fragment
#define TILE_HALFS (KB_ * FRAG_HALFS)   // 6144 halfs = 12 KB per A s-tile

// workspace layout (halfs):
//   fqB: [B_*Q_][TT_TILES][KB_][FRAG_HALFS]  (B-matrix fragments)
//   fsA: [B_*N_][ST_TILES][KB_][FRAG_HALFS]  (A-matrix fragments)
#define FQ_HALFS ((size_t)B_ * Q_ * TT_TILES * KB_ * FRAG_HALFS)   // 11,980,800
#define FS_HALFS ((size_t)B_ * N_ * ST_TILES * KB_ * FRAG_HALFS)   //  3,809,280
// total workspace: (FQ_HALFS + FS_HALFS)*2 bytes ~= 30.2 MiB

// ---------------------------------------------------------------------------
// Stage feat_query: L2-normalize each token over c, emit f16 in WMMA B layout.
// B (32x16 f16), k order used here: lane L<16 -> col N=L, K=0..15 ; lane L+16 -> col N=L, K=16..31
// One block per (b,q,t) token; t in [0, T_PAD), pad tokens write zeros.
// ---------------------------------------------------------------------------
__global__ void __launch_bounds__(384)
stage_fq_kernel(const float* __restrict__ fq, h16* __restrict__ out) {
  const int blk = blockIdx.x;            // bq * T_PAD + t
  const int bq  = blk / T_PAD;
  const int t   = blk % T_PAD;
  const int c   = threadIdx.x;           // 384 threads, one per channel

  const int t_tile  = t >> 4;
  const int lane_lo = t & 15;
  const int kb      = c >> 5;
  const int within  = c & 31;
  const int lane    = lane_lo + ((within & 16) ? 16 : 0);
  const int pos     = within & 15;
  const size_t dst  = (((size_t)bq * TT_TILES + t_tile) * KB_ + kb) * FRAG_HALFS
                      + (size_t)lane * 16 + pos;

  if (t >= T_) { out[dst] = (h16)0.f; return; }  // whole block shares t: safe

  const float x = fq[((size_t)bq * T_ + t) * C_ + c];

  // block L2-norm reduction (12 waves)
  float ss = x * x;
  for (int off = 16; off; off >>= 1) ss += __shfl_down(ss, off, 32);
  __shared__ float wsum[12];
  __shared__ float inv_s;
  const int w = threadIdx.x >> 5, ln = threadIdx.x & 31;
  if (ln == 0) wsum[w] = ss;
  __syncthreads();
  if (threadIdx.x == 0) {
    float tot = 0.f;
    for (int i = 0; i < 12; ++i) tot += wsum[i];
    inv_s = 1.0f / fmaxf(sqrtf(tot), 1e-8f);
  }
  __syncthreads();
  out[dst] = (h16)(x * inv_s);
}

// ---------------------------------------------------------------------------
// Stage feat_shot: L2-normalize each shot token, emit f16 in WMMA A layout.
// A (16x32 f16), k order used here: lane L<16 -> row M=L, K = 0..15;
//                lane L+16 -> row M=L, K = 16..31.
// s in [0, S_PAD); pad rows duplicate the last valid row (max-neutral).
// ---------------------------------------------------------------------------
__global__ void __launch_bounds__(384)
stage_fs_kernel(const float* __restrict__ fs, h16* __restrict__ out) {
  const int blk = blockIdx.x;            // bn * S_PAD + s
  const int bn  = blk / S_PAD;
  const int s   = blk % S_PAD;
  const int c   = threadIdx.x;

  const int s_tile = s >> 4;
  const int row_lo = s & 15;
  const int kb     = c >> 5;
  const int within = c & 31;
  const int group  = within >> 3;                 // 0..3
  const int lane   = row_lo + ((group & 2) ? 16 : 0);
  const int j      = within & 15;                 // position in 16-chunk
  const size_t dst = (((size_t)bn * ST_TILES + s_tile) * KB_ + kb) * FRAG_HALFS
                     + (size_t)lane * 16 + j;

  const int src_s = (s < S_) ? s : (S_ - 1);      // clamp: duplicate last row
  const float x = fs[((size_t)bn * S_ + src_s) * C_ + c];

  float ss = x * x;
  for (int off = 16; off; off >>= 1) ss += __shfl_down(ss, off, 32);
  __shared__ float wsum[12];
  __shared__ float inv_s;
  const int w = threadIdx.x >> 5, ln = threadIdx.x & 31;
  if (ln == 0) wsum[w] = ss;
  __syncthreads();
  if (threadIdx.x == 0) {
    float tot = 0.f;
    for (int i = 0; i < 12; ++i) tot += wsum[i];
    inv_s = 1.0f / fmaxf(sqrtf(tot), 1e-8f);
  }
  __syncthreads();
  out[dst] = (h16)(x * inv_s);
}

// order-preserving float <-> uint keys for atomic max on possibly-negative data
__device__ __forceinline__ unsigned f2key(float f) {
  unsigned u = __float_as_uint(f);
  return (u & 0x80000000u) ? ~u : (u | 0x80000000u);
}
__device__ __forceinline__ float key2f(unsigned k) {
  unsigned u = (k & 0x80000000u) ? (k ^ 0x80000000u) : ~k;
  return __uint_as_float(u);
}

// async copy of one 12KB A s-tile (768 x 16B chunks) into LDS; all 256 threads
__device__ __forceinline__ void async_copy_tile(const h16* __restrict__ g,
                                                h16* lds_dst) {
#pragma unroll
  for (int j = 0; j < 3; ++j) {
    const int ch = threadIdx.x + j * 256;      // 16-byte chunk index
    __builtin_amdgcn_global_load_async_to_lds_b128(
        (g_v4i*)(g + (size_t)ch * 8),
        (l_v4i*)(lds_dst + (size_t)ch * 8),
        0, 0);
  }
}

// ---------------------------------------------------------------------------
// Token-similarity GEMM + max-over-s + mean-over-t.
// One block (8 wave32) per (b,q,n).  D[s(M) x t(N)] orientation:
// max over s = per-lane max of 8 acc VGPRs + running max over s_tiles
//            + one shfl_xor(16) to merge lane halves.
// A s-tiles are streamed once per pass through LDS via async global->LDS
// copies (double-buffered, overlapped with WMMA of the previous tile),
// shared by all 8 waves instead of 8x redundant L2 reads.
// Two uniform passes over t-tiles (wave w owns tt = w and w+8) so the
// barriers inside the s-tile stream stay block-uniform; WMMA regions are
// guarded by wave-uniform branches so EXEC is all-1s.
// The k-loop rotates the A fragment one step ahead (a_cur/a_nxt) so the
// ds_load pair of fragment kb+1 issues before the WMMA of fragment kb,
// letting the backend hide LDS latency behind the matrix op with partial
// s_wait_dscnt waits instead of dscnt==0 round-trips.
// ---------------------------------------------------------------------------
__global__ void __launch_bounds__(256)
sim_gemm_kernel(const h16* __restrict__ fqB, const h16* __restrict__ fsA,
                float* __restrict__ out_logits) {
  const int bqn  = blockIdx.x;           // (b*Q + q)*N + n  (row-major [b,q,n])
  const int ncls = bqn % N_;
  const int bq   = bqn / N_;
  const int b    = bq / Q_;
  const int bn   = b * N_ + ncls;
  const int wave = threadIdx.x >> 5;
  const int lane = threadIdx.x & 31;

  __shared__ __attribute__((aligned(32))) h16 Abuf[2][TILE_HALFS];  // 2 x 12KB
  __shared__ unsigned maxkey[T_PAD];
  __shared__ float wsum[8];

  for (int i = threadIdx.x; i < T_PAD; i += 256) maxkey[i] = 0u;  // key(min)

  const h16* __restrict__ Bbase = fqB + (size_t)bq * (TT_TILES * KB_ * FRAG_HALFS);
  const h16* __restrict__ Abase = fsA + (size_t)bn * (ST_TILES * TILE_HALFS);

  for (int rep = 0; rep < 2; ++rep) {
    const int  tt  = wave + rep * 8;
    const bool act = (tt < TT_TILES);       // wave-uniform predicate

    v16h Bf[KB_];
    float m = -3.402823466e38f;
    if (act) {                              // keep this wave's B panel in VGPRs
#pragma unroll
      for (int kb = 0; kb < KB_; ++kb)
        Bf[kb] = *(const v16h*)(Bbase + ((size_t)tt * KB_ + kb) * FRAG_HALFS
                                + (size_t)lane * 16);
    }

    __syncthreads();                        // prior readers of Abuf[0] done
    async_copy_tile(Abase, &Abuf[0][0]);    // prefetch s-tile 0

    for (int st = 0; st < ST_TILES; ++st) {
      const int cur = st & 1;
      asm volatile("s_wait_asynccnt 0x0" ::: "memory");  // our copies landed
      __syncthreads();                      // everyone's copies visible
      if (st + 1 < ST_TILES)                // overlap next copy with WMMAs
        async_copy_tile(Abase + (size_t)(st + 1) * TILE_HALFS,
                        &Abuf[cur ^ 1][0]);
      if (act) {
        v8f acc = {};
        const h16* Ab = &Abuf[cur][lane * 16];
        v16h a_cur = *(const v16h*)(Ab);    // fragment kb = 0
        v16h a_nxt = a_cur;
#pragma unroll
        for (int kb = 0; kb < KB_; ++kb) {
          if (kb + 1 < KB_)                 // issue loads for kb+1 first
            a_nxt = *(const v16h*)(Ab + (kb + 1) * FRAG_HALFS);
          acc = __builtin_amdgcn_wmma_f32_16x16x32_f16(
              false, a_cur, false, Bf[kb], (short)0, acc, false, false);
          a_cur = a_nxt;
        }
        float lm = acc[0];
#pragma unroll
        for (int i = 1; i < 8; ++i) lm = fmaxf(lm, acc[i]);
        m = fmaxf(m, lm);                   // running max over s tiles
      }
    }
    if (act) {
      m = fmaxf(m, __shfl_xor(m, 16, 32));  // merge rows M and M+8
      if (lane < 16)
        atomicMax(&maxkey[tt * 16 + lane], f2key(m));   // ds_max_u32
    }
  }
  __syncthreads();

  // mean over valid t columns
  float v = 0.f;
  for (int t = threadIdx.x; t < T_; t += 256) v += key2f(maxkey[t]);
  for (int off = 16; off; off >>= 1) v += __shfl_down(v, off, 32);
  if (lane == 0) wsum[wave] = v;
  __syncthreads();
  if (threadIdx.x == 0) {
    float tot = 0.f;
    for (int i = 0; i < 8; ++i) tot += wsum[i];
    out_logits[bqn] = tot / (float)T_;
  }
}

// ---------------------------------------------------------------------------
// Global-feature classifier branch: 10 * cos(x_query, mean_k(x_shot))
// One block (4 wave32) per (b,q,n).
// ---------------------------------------------------------------------------
__global__ void __launch_bounds__(128)
cls_kernel(const float* __restrict__ xs, const float* __restrict__ xq,
           float* __restrict__ out) {
  const int idx  = blockIdx.x;           // (b*Q + q)*N + n
  const int ncls = idx % N_;
  const int bq   = idx / N_;
  const int b    = bq / Q_;

  float d = 0.f, nq = 0.f, np = 0.f;
  for (int c = threadIdx.x; c < C_; c += 128) {
    const float q = xq[(size_t)bq * C_ + c];
    float p = 0.f;
#pragma unroll
    for (int kk = 0; kk < K_; ++kk)
      p += xs[(((size_t)b * N_ + ncls) * K_ + kk) * C_ + c];
    p *= (1.0f / (float)K_);
    d += q * p;  nq += q * q;  np += p * p;
  }
  for (int off = 16; off; off >>= 1) {
    d  += __shfl_down(d,  off, 32);
    nq += __shfl_down(nq, off, 32);
    np += __shfl_down(np, off, 32);
  }
  __shared__ float sd[4], snq[4], snp[4];
  const int w = threadIdx.x >> 5, ln = threadIdx.x & 31;
  if (ln == 0) { sd[w] = d; snq[w] = nq; snp[w] = np; }
  __syncthreads();
  if (threadIdx.x == 0) {
    float D = 0.f, NQ = 0.f, NP = 0.f;
    for (int i = 0; i < 4; ++i) { D += sd[i]; NQ += snq[i]; NP += snp[i]; }
    out[B_ * Q_ * N_ + idx] =
        10.0f * D / (fmaxf(sqrtf(NQ), 1e-12f) * fmaxf(sqrtf(NP), 1e-12f));
  }
}

extern "C" void kernel_launch(void* const* d_in, const int* in_sizes, int n_in,
                              void* d_out, int out_size, void* d_ws, size_t ws_size,
                              hipStream_t stream) {
  (void)in_sizes; (void)n_in; (void)out_size; (void)ws_size;
  const float* feat_shot  = (const float*)d_in[0];  // [2,5,5,196,384]
  const float* feat_query = (const float*)d_in[1];  // [2,75,196,384]
  const float* x_shot     = (const float*)d_in[2];  // [2,5,5,384]
  const float* x_query    = (const float*)d_in[3];  // [2,75,384]
  float* out = (float*)d_out;                       // [750 logits | 750 cls]

  h16* fqB = (h16*)d_ws;
  h16* fsA = fqB + FQ_HALFS;

  stage_fq_kernel<<<B_ * Q_ * T_PAD, 384, 0, stream>>>(feat_query, fqB);
  stage_fs_kernel<<<B_ * N_ * S_PAD, 384, 0, stream>>>(feat_shot, fsA);
  sim_gemm_kernel<<<B_ * Q_ * N_, 256, 0, stream>>>(fqB, fsA, out);
  cls_kernel<<<B_ * Q_ * N_, 128, 0, stream>>>(x_shot, x_query, out);
}
